// Head_24481313587327
// MI455X (gfx1250) — matrix operand order repaired and checked by reading, not compiled, not run on hardware
//
#include <hip/hip_runtime.h>
#include <hip/hip_bf16.h>

// ---------------------------------------------------------------------------
// Fused attention head for MI455X (gfx1250, wave32, WMMA f16 -> f32)
//   q = Q@Wq+bq ; k = K@Wk+bk ; v = V@Wv+bv        (f16 intermediates)
//   scores[b,i,j] = (k_i . q_j)/8, causal j<=i, softmax over j, @ v
// ---------------------------------------------------------------------------

typedef _Float16 v16h __attribute__((ext_vector_type(16)));
typedef _Float16 v8h  __attribute__((ext_vector_type(8)));
typedef float    v8f  __attribute__((ext_vector_type(8)));
typedef float    f32x4 __attribute__((ext_vector_type(4)));

#define B_ 4
#define S_ 2048
#define E_ 1024
#define H_ 64

union HFrag { v16h v; v8h h[2]; };

__device__ __forceinline__ v8f wmma_f16(v16h a, v16h b, v8f c) {
  // (neg_a, A, neg_b, B, c_mod, C, reuse_a, reuse_b)
  return __builtin_amdgcn_wmma_f32_16x16x32_f16(false, a, false, b, (short)0, c,
                                                false, false);
}

__device__ __forceinline__ v8f vzero8() {
  v8f z = {0.f, 0.f, 0.f, 0.f, 0.f, 0.f, 0.f, 0.f};
  return z;
}

// A-fragment from f32 row: ISA A-layout, lane half selects 8-half runs at
// K = base+{0..7} and base+{16..23} (base = halfSel*8).
__device__ __forceinline__ HFrag load_a_f32(const float* __restrict__ arow,
                                            int kbase) {
  f32x4 f0 = *(const f32x4*)(arow + kbase);
  f32x4 f1 = *(const f32x4*)(arow + kbase + 4);
  f32x4 f2 = *(const f32x4*)(arow + kbase + 16);
  f32x4 f3 = *(const f32x4*)(arow + kbase + 20);
  HFrag a;
  for (int e = 0; e < 4; ++e) {
    a.h[0][e]     = (_Float16)f0[e];
    a.h[0][e + 4] = (_Float16)f1[e];
    a.h[1][e]     = (_Float16)f2[e];
    a.h[1][e + 4] = (_Float16)f3[e];
  }
  return a;
}

// B-fragment from f32 row: contiguous K order (proj_qk's A matches), 16 K per lane
// (lanes 0-15: K=0-15, lanes 16-31: K=16-31) -> straight conversion.
__device__ __forceinline__ HFrag load_b_f32(const float* __restrict__ brow,
                                            int kbase) {
  f32x4 f0 = *(const f32x4*)(brow + kbase);
  f32x4 f1 = *(const f32x4*)(brow + kbase + 4);
  f32x4 f2 = *(const f32x4*)(brow + kbase + 8);
  f32x4 f3 = *(const f32x4*)(brow + kbase + 12);
  HFrag b;
  for (int e = 0; e < 4; ++e) {
    b.h[0][e]     = (_Float16)f0[e];
    b.h[0][e + 4] = (_Float16)f1[e];
    b.h[1][e]     = (_Float16)f2[e];
    b.h[1][e + 4] = (_Float16)f3[e];
  }
  return b;
}

// ---------------------------------------------------------------------------
// Kernel 0: transpose+convert the three [E,H] fp32 weight matrices into
// f16 Wt[m][h][e]  (m = 0:Wq, 1:Wk, 2:Wv).
// ---------------------------------------------------------------------------
__global__ __launch_bounds__(256) void prep_weights(const float* __restrict__ Wq,
                                                    const float* __restrict__ Wk,
                                                    const float* __restrict__ Wv,
                                                    _Float16* __restrict__ Wt) {
  int idx = blockIdx.x * blockDim.x + threadIdx.x;   // 3*H_*E_ threads
  int m = idx / (H_ * E_);
  int r = idx % (H_ * E_);
  int h = r / E_;
  int e = r % E_;
  const float* W = (m == 0) ? Wq : ((m == 1) ? Wk : Wv);
  Wt[(size_t)m * H_ * E_ + (size_t)h * E_ + e] = (_Float16)W[(size_t)e * H_ + h];
}

// ---------------------------------------------------------------------------
// Kernel 1a: q/k projections, computed TRANSPOSED (D rows = h, cols = s):
//   D[h][s] = sum_e Wt[h][e] * X[s][e]   (A = Wt rows, B = X rows as columns)
// so each lane's 8 accumulator rows are 8 consecutive h -> one b128 store
// per tile into row-major dst[s][h]. Grid: x = (B*S)/256, y = 2 (q, k).
// 8 waves/block; each wave: 64h x 32s tile, 8 WMMAs per 16 b128 loads.
// ---------------------------------------------------------------------------
__global__ __launch_bounds__(256) void proj_qk(const float* __restrict__ Q,
                                               const float* __restrict__ K,
                                               const float* __restrict__ bq,
                                               const float* __restrict__ bk,
                                               const _Float16* __restrict__ Wt,
                                               _Float16* __restrict__ qp,
                                               _Float16* __restrict__ kp) {
  const int lane    = threadIdx.x & 31;
  const int wave    = threadIdx.x >> 5;
  const int proj    = blockIdx.y;
  const int s0      = blockIdx.x * 256 + wave * 32;   // row in [0, B*S)
  const int halfSel = lane >> 4;
  const int lrow    = lane & 15;

  const float* src      = (proj == 0) ? Q : K;
  const float* bias     = (proj == 0) ? bq : bk;
  const _Float16* W     = Wt + (size_t)proj * H_ * E_;
  _Float16* dst         = (proj == 0) ? qp : kp;

  v8f acc[2][4];                                      // [row-tile][h-tile]
  for (int rt = 0; rt < 2; ++rt)
    for (int ht = 0; ht < 4; ++ht) acc[rt][ht] = vzero8();

  const float* xrow0 = src + (size_t)(s0 + lrow) * E_;
  const float* xrow1 = src + (size_t)(s0 + 16 + lrow) * E_;

  for (int k0 = 0; k0 < E_; k0 += 32) {
    // B fragments: streaming X rows (the HBM traffic), contiguous 16 floats.
    HFrag b0 = load_b_f32(xrow0, k0 + halfSel * 16);
    HFrag b1 = load_b_f32(xrow1, k0 + halfSel * 16);
    // A fragments: L2-resident transposed weights, shared by both B's.
    for (int ht = 0; ht < 4; ++ht) {
      HFrag a;
      const _Float16* wrow =
          W + (size_t)(ht * 16 + lrow) * E_ + k0 + halfSel * 16;
      a.h[0] = *(const v8h*)(wrow);
      a.h[1] = *(const v8h*)(wrow + 8);
      acc[0][ht] = wmma_f16(a.v, b0.v, acc[0][ht]);
      acc[1][ht] = wmma_f16(a.v, b1.v, acc[1][ht]);
    }
  }

  // Epilogue: bias + pack 8 consecutive h per lane -> coalesced b128 stores.
  for (int rt = 0; rt < 2; ++rt) {
    const int s = s0 + rt * 16 + lrow;                // this lane's column
    for (int ht = 0; ht < 4; ++ht) {
      const int h0 = ht * 16 + halfSel * 8;           // first of 8 h values
      f32x4 bv0 = *(const f32x4*)(bias + h0);
      f32x4 bv1 = *(const f32x4*)(bias + h0 + 4);
      v8h p;
      for (int v = 0; v < 4; ++v) {
        p[v]     = (_Float16)(acc[rt][ht][v]     + bv0[v]);
        p[v + 4] = (_Float16)(acc[rt][ht][v + 4] + bv1[v]);
      }
      *(v8h*)(dst + (size_t)s * H_ + h0) = p;
    }
  }
}

// ---------------------------------------------------------------------------
// Kernel 1b: v projection, normal orientation (D rows = s, cols = h):
// each lane's 8 accumulator rows are 8 consecutive s -> one b128 store per
// tile into transposed vT[b][h][s]. Grid: x = (B*S)/256.
// ---------------------------------------------------------------------------
__global__ __launch_bounds__(256) void proj_v(const float* __restrict__ V,
                                              const float* __restrict__ bv,
                                              const _Float16* __restrict__ Wt,
                                              _Float16* __restrict__ vT) {
  const int lane    = threadIdx.x & 31;
  const int wave    = threadIdx.x >> 5;
  const int r0      = blockIdx.x * 256 + wave * 32;   // row in [0, B*S)
  const int halfSel = lane >> 4;
  const int lrow    = lane & 15;

  const _Float16* W = Wt + (size_t)2 * H_ * E_;

  v8f acc[2][4];                                      // [row-tile][h-tile]
  for (int rt = 0; rt < 2; ++rt)
    for (int nt = 0; nt < 4; ++nt) acc[rt][nt] = vzero8();

  const float* arow0 = V + (size_t)(r0 + lrow) * E_;
  const float* arow1 = V + (size_t)(r0 + 16 + lrow) * E_;

  for (int k0 = 0; k0 < E_; k0 += 32) {
    const int kbase = k0 + halfSel * 8;
    HFrag a0 = load_a_f32(arow0, kbase);
    HFrag a1 = load_a_f32(arow1, kbase);
    for (int nt = 0; nt < 4; ++nt) {
      HFrag bf;
      const _Float16* wcol =
          W + (size_t)(nt * 16 + lrow) * E_ + k0 + halfSel * 8;
      bf.h[0] = *(const v8h*)(wcol);
      bf.h[1] = *(const v8h*)(wcol + 16);
      acc[0][nt] = wmma_f16(a0.v, bf.v, acc[0][nt]);
      acc[1][nt] = wmma_f16(a1.v, bf.v, acc[1][nt]);
    }
  }

  // Epilogue: 8 consecutive s per lane -> coalesced b128 stores into vT.
  const int bb = r0 >> 11;                            // batch (never straddled)
  const int sbase = r0 & (S_ - 1);
  for (int rt = 0; rt < 2; ++rt) {
    for (int nt = 0; nt < 4; ++nt) {
      const int h  = nt * 16 + lrow;                  // this lane's column
      const float bval = bv[h];
      const int sA = sbase + rt * 16 + halfSel * 8;   // first of 8 s values
      v8h p;
      for (int v = 0; v < 8; ++v)
        p[v] = (_Float16)(acc[rt][nt][v] + bval);
      *(v8h*)(vT + ((size_t)bb * H_ + h) * S_ + sA) = p;
    }
  }
}

// ---------------------------------------------------------------------------
// Kernel 2: causal flash attention.
//   scores rows = kproj rows (i), scores cols = qproj rows (j).
// Grid: x = S/128, y = B. 8 waves/block, each wave one 16-row i-tile.
// All operands are L2-resident f16 (~3 MB inside the 192 MB L2).
// ---------------------------------------------------------------------------
__global__ __launch_bounds__(256) void attn_kernel(const _Float16* __restrict__ qp,
                                                   const _Float16* __restrict__ kp,
                                                   const _Float16* __restrict__ vT,
                                                   float* __restrict__ out) {
  __shared__ __align__(16) _Float16 lds[8][16 * 32];  // per-wave P staging

  const int lane    = threadIdx.x & 31;
  const int wave    = threadIdx.x >> 5;
  const int b       = blockIdx.y;
  const int i0      = blockIdx.x * 128 + wave * 16;
  const int halfSel = lane >> 4;
  const int lrow    = lane & 15;

  const _Float16* qb = qp + (size_t)b * S_ * H_;
  const _Float16* kb = kp + (size_t)b * S_ * H_;
  const _Float16* vb = vT + (size_t)b * H_ * S_;

  // A fragments for the 16 k-rows of this tile (2 k-steps over H=64).
  HFrag ka[2];
  {
    const _Float16* krow = kb + (size_t)(i0 + lrow) * H_;
    for (int ks = 0; ks < 2; ++ks) {
      const int kb0 = ks * 32 + halfSel * 8;
      ka[ks].h[0] = *(const v8h*)(krow + kb0);
      ka[ks].h[1] = *(const v8h*)(krow + kb0 + 16);
    }
  }

  v8f O[4];
  for (int nt = 0; nt < 4; ++nt) O[nt] = vzero8();
  float m_[8], l_[8];
  for (int v = 0; v < 8; ++v) { m_[v] = -INFINITY; l_[v] = 0.f; }

  const int iMax = i0 + 15;
  for (int j0 = 0; j0 <= iMax; j0 += 32) {           // causal chunk skip
    // ---- scores: two 16x16 j-tiles, each accumulating 2 WMMA k-steps
    v8f sc[2];
    for (int jt = 0; jt < 2; ++jt) {
      HFrag qf[2];
      const _Float16* qrow = qb + (size_t)(j0 + jt * 16 + lrow) * H_;
      for (int ks = 0; ks < 2; ++ks) {
        const int kb0 = ks * 32 + halfSel * 8;       // same k order as A: 8-runs at +0, +16
        qf[ks].h[0] = *(const v8h*)(qrow + kb0);
        qf[ks].h[1] = *(const v8h*)(qrow + kb0 + 16);
      }
      v8f z = vzero8();
      z = wmma_f16(ka[0].v, qf[0].v, z);
      z = wmma_f16(ka[1].v, qf[1].v, z);
      sc[jt] = z;
    }

    // ---- scale, causal mask, per-row chunk max (reduce across 16-lane half)
    float chmax[8];
    for (int v = 0; v < 8; ++v) {
      const int irow = i0 + v + 8 * halfSel;
      float best = -INFINITY;
      for (int jt = 0; jt < 2; ++jt) {
        const int j = j0 + jt * 16 + lrow;
        float s = sc[jt][v] * 0.125f;                // 1/sqrt(64)
        if (j > irow) s = -INFINITY;
        sc[jt][v] = s;
        best = fmaxf(best, s);
      }
      for (int d = 1; d < 16; d <<= 1)
        best = fmaxf(best, __shfl_xor(best, d, 32));
      chmax[v] = best;
    }

    // ---- online softmax update
    float alpha[8];
    for (int v = 0; v < 8; ++v) {
      const float mnew = fmaxf(m_[v], chmax[v]);
      alpha[v] = __expf(m_[v] - mnew);
      m_[v] = mnew;
      float ps = 0.f;
      for (int jt = 0; jt < 2; ++jt) {
        const float p = __expf(sc[jt][v] - mnew);
        sc[jt][v] = p;
        ps += p;
      }
      for (int d = 1; d < 16; d <<= 1)
        ps += __shfl_xor(ps, d, 32);
      l_[v] = l_[v] * alpha[v] + ps;
    }
    for (int nt = 0; nt < 4; ++nt)
      for (int v = 0; v < 8; ++v)
        O[nt][v] *= alpha[v];

    // ---- P: C-layout -> A-layout via per-wave LDS (f16, row-major 16x32)
    for (int v = 0; v < 8; ++v) {
      const int row = v + 8 * halfSel;
      lds[wave][row * 32 + lrow]      = (_Float16)sc[0][v];
      lds[wave][row * 32 + 16 + lrow] = (_Float16)sc[1][v];
    }
    asm volatile("s_wait_dscnt 0" ::: "memory");     // same-wave LDS RAW

    HFrag pa;
    {
      const _Float16* prow = &lds[wave][lrow * 32 + halfSel * 8];
      pa.h[0] = *(const v8h*)(prow);
      pa.h[1] = *(const v8h*)(prow + 16);
    }

    // ---- O += P @ V  (V B-fragments contiguous thanks to vT layout)
    for (int nt = 0; nt < 4; ++nt) {
      HFrag vf;
      const _Float16* vcol =
          vb + (size_t)(nt * 16 + lrow) * S_ + j0 + halfSel * 8;
      vf.h[0] = *(const v8h*)(vcol);
      vf.h[1] = *(const v8h*)(vcol + 16);
      O[nt] = wmma_f16(pa.v, vf.v, O[nt]);
    }
  }

  // ---- normalize + store fp32 output [B,S,H]
  for (int nt = 0; nt < 4; ++nt) {
    for (int v = 0; v < 8; ++v) {
      const int s = i0 + v + 8 * halfSel;
      out[((size_t)b * S_ + s) * H_ + nt * 16 + lrow] = O[nt][v] / l_[v];
    }
  }
}

// ---------------------------------------------------------------------------
extern "C" void kernel_launch(void* const* d_in, const int* in_sizes, int n_in,
                              void* d_out, int out_size, void* d_ws,
                              size_t ws_size, hipStream_t stream) {
  (void)in_sizes; (void)n_in; (void)out_size; (void)ws_size;
  const float* Q  = (const float*)d_in[0];
  const float* V  = (const float*)d_in[1];
  const float* K  = (const float*)d_in[2];
  const float* Wq = (const float*)d_in[3];
  const float* bq = (const float*)d_in[4];
  const float* Wv = (const float*)d_in[5];
  const float* bv = (const float*)d_in[6];
  const float* Wk = (const float*)d_in[7];
  const float* bk = (const float*)d_in[8];
  float* out = (float*)d_out;

  // Workspace layout (f16): Wt[3*H*E] | qp[B*S*H] | kp[B*S*H] | vT[B*H*S]
  _Float16* Wt = (_Float16*)d_ws;
  _Float16* qp = Wt + (size_t)3 * H_ * E_;
  _Float16* kp = qp + (size_t)B_ * S_ * H_;
  _Float16* vT = kp + (size_t)B_ * S_ * H_;
  // total: (3*64*1024 + 3*4*2048*64) * 2 B ~= 3.4 MB of d_ws

  prep_weights<<<dim3((3 * H_ * E_) / 256), dim3(256), 0, stream>>>(Wq, Wk, Wv, Wt);
  proj_qk<<<dim3((B_ * S_) / 256, 2), dim3(256), 0, stream>>>(
      Q, K, bq, bk, Wt, qp, kp);
  proj_v<<<dim3((B_ * S_) / 256), dim3(256), 0, stream>>>(V, bv, Wt, vT);
  attn_kernel<<<dim3(S_ / 128, B_), dim3(256), 0, stream>>>(qp, kp, vT, out);
}
